// GenerativeModel_71648644432473
// MI455X (gfx1250) — matrix operand order repaired and checked by reading, not compiled, not run on hardware
//
#include <hip/hip_runtime.h>
#include <hip/hip_bf16.h>

#define IM   64
#define NPIX (IM*IM)
#define HID  64
#define WROW 72   // padded LDS row stride (halves) to spread banks; keeps 16B alignment

typedef __attribute__((ext_vector_type(16))) _Float16 v16h;
typedef __attribute__((ext_vector_type(8)))  _Float16 v8h;
typedef __attribute__((ext_vector_type(8)))  float    v8f;

__device__ __forceinline__ float sigmoidf_(float x) { return 1.0f / (1.0f + __expf(-x)); }

__global__ __launch_bounds__(256, 1)
void genmodel_kernel(const int*   __restrict__ program_id,
                     const int*   __restrict__ shape_ids,
                     const float* __restrict__ raw_positions,
                     const float* __restrict__ w0, const float* __restrict__ b0,
                     const float* __restrict__ w1, const float* __restrict__ b1,
                     const float* __restrict__ w2, const float* __restrict__ b2,
                     const float* __restrict__ w3, const float* __restrict__ b3,
                     const float* __restrict__ lm,
                     float*       __restrict__ out)
{
    // LDS: transposed f16 weights (per s), per-wave activation staging, p0 image buffer
    __shared__ _Float16 wt1[2][HID * WROW];   // [s][o*WROW + i]
    __shared__ _Float16 wt2[2][HID * WROW];
    __shared__ float w0s[2][HID], b0s[2][HID], b1s[2][HID], b2s[2][HID], w3s[2][HID];
    __shared__ float p0buf[NPIX];
    __shared__ _Float16 stage[8][16 * WROW];  // per-wave [m][k] f16 staging (D->A shuffle)
    __shared__ float osum[8][16];             // per-wave reduced layer-3 outputs

    const int b    = blockIdx.x;
    const int tid  = threadIdx.x;
    const int lane = tid & 31;
    const int wave = tid >> 5;
    const int hi   = lane >> 4;    // half-wave select (K sub-block per ISA layout)
    const int l15  = lane & 15;    // row (A: M) or column (B/D: N) index

    const int pid    = program_id[b];
    const int p_s[2] = { shape_ids[b * 2 + 0], shape_ids[b * 2 + 1] };

    // ---- stage weights into LDS (transposed, f16) ----
    for (int s = 0; s < 2; ++s) {
        const int p = p_s[s];
        const float* w1p = w1 + p * HID * HID;
        const float* w2p = w2 + p * HID * HID;
        for (int f = tid; f < HID * HID; f += 256) {
            const int i = f >> 6, o = f & 63;       // w[p][i][o]
            wt1[s][o * WROW + i] = (_Float16)w1p[f];
            wt2[s][o * WROW + i] = (_Float16)w2p[f];
        }
        if (tid < HID) {
            w0s[s][tid] = w0[p * HID + tid];
            b0s[s][tid] = b0[p * HID + tid];
            b1s[s][tid] = b1[p * HID + tid];
            b2s[s][tid] = b2[p * HID + tid];
            w3s[s][tid] = w3[p * HID + tid];
        }
    }
    __syncthreads();

#pragma unroll
    for (int s = 0; s < 2; ++s) {
        const int   p    = p_s[s];
        const float mult = __expf(lm[p]);
        const float b3v  = b3[p];
        const float px   = sigmoidf_(raw_positions[(b * 2 + s) * 2 + 0]) - 0.5f;
        const float py   = sigmoidf_(raw_positions[(b * 2 + s) * 2 + 1]) - 0.5f;

        // Per-lane layer-0 coefficients in A-fragment order:
        // element j of the 16x32 f16 A frag holds K = (j&7) + 16*(j>>3) + 8*hi (+32*q)
        v16h w0f[2], b0f[2];
#pragma unroll
        for (int q = 0; q < 2; ++q)
#pragma unroll
            for (int j = 0; j < 16; ++j) {
                const int k = q * 32 + (j & 7) + ((j >> 3) * 16) + hi * 8;
                w0f[q][j] = (_Float16)w0s[s][k];
                b0f[q][j] = (_Float16)b0s[s][k];
            }

        float b1v[4], b2v[4], w3v[4];
#pragma unroll
        for (int t = 0; t < 4; ++t) {
            b1v[t] = b1s[s][t * 16 + l15];
            b2v[t] = b2s[s][t * 16 + l15];
            w3v[t] = w3s[s][t * 16 + l15];
        }

        for (int tt = 0; tt < 32; ++tt) {
            const int tile = tt * 8 + wave;           // 256 M-tiles / 8 waves
            const int pix  = tile * 16 + l15;         // both half-waves mirror M rows
            const int x    = pix & 63, y = pix >> 6;
            const float cx = -1.0f + (2.0f / 63.0f) * (float)x;
            const float cy = -1.0f + (2.0f / 63.0f) * (float)y;
            const float dx = cx - px, dy = cy - py;
            const float theta = atan2f(dy, dx);
            const float rdist = sqrtf(dx * dx + dy * dy);

            // ---- layer 0: outer product -> f16 A fragments in registers ----
            const _Float16 th = (_Float16)theta;
            const _Float16 zh = (_Float16)0.0f;
            v16h a0[2];
#pragma unroll
            for (int q = 0; q < 2; ++q)
#pragma unroll
                for (int j = 0; j < 16; ++j) {
                    _Float16 h = th * w0f[q][j] + b0f[q][j];
                    a0[q][j] = h > zh ? h : zh;
                }

            // ---- layer 1: (16x64) @ (64x64) via WMMA ----
            v8f acc[4];
#pragma unroll
            for (int t = 0; t < 4; ++t) {
                v8f c = {};
#pragma unroll
                for (int q = 0; q < 2; ++q) {
                    const int boff = (t * 16 + l15) * WROW + q * 32 + hi * 8;
                    const v8h blo = *(const v8h*)&wt1[s][boff];
                    const v8h bhi = *(const v8h*)&wt1[s][boff + 16];
                    v16h bw;
#pragma unroll
                    for (int j = 0; j < 8; ++j) { bw[j] = blo[j]; bw[j + 8] = bhi[j]; }
                    c = __builtin_amdgcn_wmma_f32_16x16x32_f16(
                            false, a0[q], false, bw, (short)0, c, false, false);
                }
                acc[t] = c;
            }

            // bias + relu, stage D-layout -> [m][k] f16 in per-wave LDS
#pragma unroll
            for (int t = 0; t < 4; ++t)
#pragma unroll
                for (int r = 0; r < 8; ++r) {
                    const float h = fmaxf(acc[t][r] + b1v[t], 0.0f);
                    stage[wave][(r + hi * 8) * WROW + t * 16 + l15] = (_Float16)h;
                }
            __builtin_amdgcn_wave_barrier();

            // reload as A fragments (the required cross-lane shuffle)
            v16h a1[2];
#pragma unroll
            for (int q = 0; q < 2; ++q) {
                const int base = l15 * WROW + q * 32 + hi * 8;
                const v8h lo8 = *(const v8h*)&stage[wave][base];
                const v8h hi8 = *(const v8h*)&stage[wave][base + 16];
#pragma unroll
                for (int j = 0; j < 8; ++j) { a1[q][j] = lo8[j]; a1[q][j + 8] = hi8[j]; }
            }

            // ---- layer 2 WMMA + bias/relu fused into layer-3 dot partials ----
            float psum[8];
#pragma unroll
            for (int r = 0; r < 8; ++r) psum[r] = 0.0f;
#pragma unroll
            for (int t = 0; t < 4; ++t) {
                v8f c = {};
#pragma unroll
                for (int q = 0; q < 2; ++q) {
                    const int boff = (t * 16 + l15) * WROW + q * 32 + hi * 8;
                    const v8h blo = *(const v8h*)&wt2[s][boff];
                    const v8h bhi = *(const v8h*)&wt2[s][boff + 16];
                    v16h bw;
#pragma unroll
                    for (int j = 0; j < 8; ++j) { bw[j] = blo[j]; bw[j + 8] = bhi[j]; }
                    c = __builtin_amdgcn_wmma_f32_16x16x32_f16(
                            false, a1[q], false, bw, (short)0, c, false, false);
                }
#pragma unroll
                for (int r = 0; r < 8; ++r) {
                    const float h = fmaxf(c[r] + b2v[t], 0.0f);
                    psum[r] += h * w3v[t];
                }
            }

            // reduce the N dimension across the 16 lanes of each half-wave
#pragma unroll
            for (int r = 0; r < 8; ++r) {
                float v = psum[r];
                v += __shfl_xor(v, 1, 32);
                v += __shfl_xor(v, 2, 32);
                v += __shfl_xor(v, 4, 32);
                v += __shfl_xor(v, 8, 32);
                psum[r] = v;
            }
#pragma unroll
            for (int r = 0; r < 8; ++r)
                if (l15 == r) osum[wave][hi * 8 + r] = psum[r];
            __builtin_amdgcn_wave_barrier();

            // ---- final pointwise math; both halves mirror pixel m = l15 ----
            const float o     = osum[wave][l15] + b3v;
            const float logit = mult * (__expf(o) - rdist);
            const float prob  = sigmoidf_(logit);

            if (s == 0) {
                if (hi == 0) p0buf[pix] = prob;
            } else {
                const float p0 = p0buf[pix];
                float comb = (pid == 2) ? (p0 - prob) : (p0 + prob);
                comb = fminf(fmaxf(comb, 0.0f), 1.0f);
                const float res = (pid == 0) ? p0 : comb;
                if (hi == 0) out[b * NPIX + pix] = res;
            }
        }
        __builtin_amdgcn_wave_barrier();
    }
}

extern "C" void kernel_launch(void* const* d_in, const int* in_sizes, int n_in,
                              void* d_out, int out_size, void* d_ws, size_t ws_size,
                              hipStream_t stream) {
    (void)in_sizes; (void)n_in; (void)out_size; (void)d_ws; (void)ws_size;
    genmodel_kernel<<<256, 256, 0, stream>>>(
        (const int*)d_in[0],    // program_id
        (const int*)d_in[1],    // shape_ids
        (const float*)d_in[2],  // raw_positions
        (const float*)d_in[3],  // w0
        (const float*)d_in[4],  // b0
        (const float*)d_in[5],  // w1
        (const float*)d_in[6],  // b1
        (const float*)d_in[7],  // w2
        (const float*)d_in[8],  // b2
        (const float*)d_in[9],  // w3
        (const float*)d_in[10], // b3
        (const float*)d_in[11], // logit_multipliers_raw
        (float*)d_out);
}
